// BertSelfAttention_59219009077811
// MI455X (gfx1250) — matrix operand order repaired and checked by reading, not compiled, not run on hardware
//
#include <hip/hip_runtime.h>
#include <hip/hip_bf16.h>

#define B_  2
#define S_  2048
#define H_  1024
#define NH_ 16
#define DH_ 64

typedef __attribute__((ext_vector_type(16))) __bf16 v16bf;
typedef __attribute__((ext_vector_type(8)))  float  v8f;
typedef __attribute__((ext_vector_type(4)))  unsigned u32x4;
typedef __attribute__((ext_vector_type(8)))  int      i32x8;

union Frag16 { unsigned u[8]; v16bf v; };

__device__ inline v8f wmma_bf16(v16bf a, v16bf b, v8f c) {
  // D(f32 16x16) = A(bf16 16x32) * B(bf16 32x16) + C
  return __builtin_amdgcn_wmma_f32_16x16x32_bf16(false, a, false, b, (short)0, c, false, false);
}

// ---- CDNA5 async global->LDS copy (ASYNCcnt-tracked), via inline asm -------
__device__ inline void async_ld16(void* lds_ptr, const void* gptr) {
  // Flat pointers to LDS carry the LDS byte offset in bits [31:0] (ISA 10.2).
  unsigned lds_off = (unsigned)(size_t)lds_ptr;
  asm volatile("global_load_async_to_lds_b128 %0, %1, off"
               :: "v"(lds_off), "v"(gptr)
               : "memory");
}
__device__ inline void wait_async0() {
  asm volatile("s_wait_asynccnt 0x0" ::: "memory");
}

// ---- CDNA5 Tensor Data Mover: 64x64 bf16 tile, global->LDS ----------------
// D# describes a 2D tile: tile_dim0=64 elems/row (128B), tile_dim1=64 rows,
// tensor_dim0_stride=64 elems.  LDS padding: after every 32 DWORDs (128B)
// insert 4 DWORDs (16B) -> LDS row pitch 144B, matching [64][72] arrays.
__device__ inline void tdm_load_64x64_bf16(unsigned lds_off, const void* gaddr) {
  unsigned long long ga = (unsigned long long)gaddr;
  u32x4 g0;
  g0[0] = 1u;                                   // count=1 (valid descriptor)
  g0[1] = lds_off;                              // lds_addr (bytes)
  g0[2] = (unsigned)ga;                         // global_addr[31:0]
  g0[3] = (unsigned)(ga >> 32) | 0x80000000u;   // global_addr[56:32] | type=2
  i32x8 g1;
  g1[0] = (1 << 16) | (1 << 20) | (4 << 22) | (3 << 25);
          // data_size=1(2B) | pad_enable | pad_interval=32dw | pad_amount=4dw
  g1[1] = (64 << 16);     // tensor_dim0[15:0] in bits[63:48]
  g1[2] = (2048 << 16);   // tensor_dim0 hi=0 | tensor_dim1 lo16 = 2048
  g1[3] = (64 << 16);     // tensor_dim1 hi=0 | tile_dim0 = 64
  g1[4] = 64;             // tile_dim1 = 64, tile_dim2 = 0
  g1[5] = 64;             // tensor_dim0_stride lo32 = 64
  g1[6] = 0;              // stride hi | tensor_dim1_stride lo (unused, 2D)
  g1[7] = 0;
  asm volatile("tensor_load_to_lds %0, %1"
               :: "s"(g0), "s"(g1)
               : "memory");
}

// A-matrix 16x32 bf16 fragment from row-major tile (ld = leading dim, elems).
// lane holds row M=lane%16; lanes0-15: K=0..15 ; lanes16-31: K=16..31
__device__ inline v16bf load_A_frag(const __bf16* src, int ld, int lane) {
  const int m = lane & 15;
  const int kbase = (lane < 16) ? 0 : 16;
  const __bf16* row = src + m * ld;
  Frag16 f;
#pragma unroll
  for (int j = 0; j < 8; ++j) {
    int kk = kbase + 2 * j;
    f.u[j] = *(const unsigned*)(row + kk);
  }
  return f.v;
}

// B-matrix 32x16 bf16 fragment where B[k][n] is stored TRANSPOSED: src[n][k] row-major.
__device__ inline v16bf load_Bt_frag(const __bf16* src, int ld, int lane) {
  const int n = lane & 15;
  const int kbase = (lane < 16) ? 0 : 16;
  const __bf16* row = src + n * ld;
  Frag16 f;
#pragma unroll
  for (int j = 0; j < 8; ++j)
    f.u[j] = *(const unsigned*)(row + kbase + 2 * j);
  return f.v;
}

// B-matrix 32x16 bf16 fragment from natural [k][n] row-major storage.
__device__ inline v16bf load_B_frag(const __bf16* src, int ld, int lane) {
  const int n = lane & 15;
  const int kbase = (lane < 16) ? 0 : 16;
  v16bf b;
#pragma unroll
  for (int j = 0; j < 16; ++j)
    b[j] = src[(kbase + j) * ld + n];
  return b;
}

// ---------------------------------------------------------------------------
// Kernel 1: fused QKV projection.  out = x @ W^T + b -> bf16 [B, NH, S, DH].
// Block tile 128(M) x 64(N), K-step 32, 8 waves, register double-buffered
// global staging so loads overlap WMMA.
// ---------------------------------------------------------------------------
__global__ __launch_bounds__(256) void qkv_kernel(
    const float* __restrict__ X,
    const float* __restrict__ Wq, const float* __restrict__ bq,
    const float* __restrict__ Wk, const float* __restrict__ bk,
    const float* __restrict__ Wv, const float* __restrict__ bv,
    __bf16* __restrict__ Qh, __bf16* __restrict__ Kh, __bf16* __restrict__ Vh)
{
  const int t     = threadIdx.x;
  const int lane  = t & 31;
  const int wave  = t >> 5;     // 0..7
  const int waveM = wave & 3;   // 0..3 -> 32-row slab
  const int waveN = wave >> 2;  // 0..1 -> 32-col slab
  const int m0 = blockIdx.y * 128;
  const int n0 = blockIdx.x * 64;

  const float* W; const float* bias; __bf16* out;
  if (blockIdx.z == 0)      { W = Wq; bias = bq; out = Qh; }
  else if (blockIdx.z == 1) { W = Wk; bias = bk; out = Kh; }
  else                      { W = Wv; bias = bv; out = Vh; }

  __shared__ __align__(16) __bf16 As[128][40];
  __shared__ __align__(16) __bf16 Bs[64][40];

  const v8f vzero = {0.f,0.f,0.f,0.f,0.f,0.f,0.f,0.f};
  v8f acc[2][2];
  acc[0][0] = vzero; acc[0][1] = vzero; acc[1][0] = vzero; acc[1][1] = vzero;

  float4 xa[4], xb[2];

  auto load_stage = [&](int k0) {
#pragma unroll
    for (int ii = 0; ii < 4; ++ii) {          // X tile: 128 rows x 8 float4
      int i = t + ii * 256;
      int row = i >> 3, c4 = (i & 7) * 4;
      xa[ii] = *(const float4*)(X + (size_t)(m0 + row) * H_ + k0 + c4);
    }
#pragma unroll
    for (int ii = 0; ii < 2; ++ii) {          // W tile: 64 rows x 8 float4
      int i = t + ii * 256;
      int row = i >> 3, c4 = (i & 7) * 4;
      xb[ii] = *(const float4*)(W + (size_t)(n0 + row) * H_ + k0 + c4);
    }
  };
  auto store_stage = [&]() {
#pragma unroll
    for (int ii = 0; ii < 4; ++ii) {
      int i = t + ii * 256;
      int row = i >> 3, c4 = (i & 7) * 4;
      As[row][c4 + 0] = (__bf16)xa[ii].x; As[row][c4 + 1] = (__bf16)xa[ii].y;
      As[row][c4 + 2] = (__bf16)xa[ii].z; As[row][c4 + 3] = (__bf16)xa[ii].w;
    }
#pragma unroll
    for (int ii = 0; ii < 2; ++ii) {
      int i = t + ii * 256;
      int row = i >> 3, c4 = (i & 7) * 4;
      Bs[row][c4 + 0] = (__bf16)xb[ii].x; Bs[row][c4 + 1] = (__bf16)xb[ii].y;
      Bs[row][c4 + 2] = (__bf16)xb[ii].z; Bs[row][c4 + 3] = (__bf16)xb[ii].w;
    }
  };

  load_stage(0);
  store_stage();
  __syncthreads();

  for (int step = 0; step < H_ / 32; ++step) {
    const bool has_next = (step + 1) < H_ / 32;
    if (has_next) load_stage((step + 1) * 32);   // prefetch next slab -> VGPRs

    v16bf b0 = load_Bt_frag(&Bs[waveN * 32 + 0][0], 40, lane);
    v16bf b1 = load_Bt_frag(&Bs[waveN * 32 + 16][0], 40, lane);
#pragma unroll
    for (int fi = 0; fi < 2; ++fi) {
      v16bf a = load_A_frag(&As[waveM * 32 + fi * 16][0], 40, lane);
      acc[fi][0] = wmma_bf16(a, b0, acc[fi][0]);
      acc[fi][1] = wmma_bf16(a, b1, acc[fi][1]);
    }

    __syncthreads();              // all waves done reading LDS slab
    if (has_next) {
      store_stage();              // commit prefetched slab
      __syncthreads();
    }
  }

  const int hf  = lane >> 4;
  const int nlo = lane & 15;
#pragma unroll
  for (int fi = 0; fi < 2; ++fi) {
#pragma unroll
    for (int fj = 0; fj < 2; ++fj) {
      int n = n0 + waveN * 32 + fj * 16 + nlo;
      float bb = bias[n];
      int nh = n >> 6, dh = n & 63;
#pragma unroll
      for (int r = 0; r < 8; ++r) {
        int m = m0 + waveM * 32 + fi * 16 + r + hf * 8;
        int bb_i = m >> 11, ss = m & (S_ - 1);
        out[(((size_t)(bb_i * NH_ + nh)) * S_ + ss) * DH_ + dh] =
            (__bf16)(acc[fi][fj][r] + bb);
      }
    }
  }
}

// ---------------------------------------------------------------------------
// Kernel 2: flash attention per (b, nh, 64-query tile). 4 waves; each wave
// owns 16 full query rows. Q staged via async global->LDS; K/V tiles staged
// with the Tensor Data Mover (one tensor_load_to_lds per tile, wave 0 only).
// ---------------------------------------------------------------------------
__global__ __launch_bounds__(128) void attn_kernel(
    const __bf16* __restrict__ Qh, const __bf16* __restrict__ Kh,
    const __bf16* __restrict__ Vh, const float* __restrict__ amask,
    float* __restrict__ out)
{
  const int t    = threadIdx.x;
  const int lane = t & 31;
  const int w    = t >> 5;       // 0..3
  const int hf   = lane >> 4;
  const int nlo  = lane & 15;
  const int q0 = blockIdx.x * 64;
  const int nh = blockIdx.y;
  const int bz = blockIdx.z;

  __shared__ __align__(16) __bf16 Qs[64][72];
  __shared__ __align__(16) __bf16 Ks[64][72];
  __shared__ __align__(16) __bf16 Vs[64][72];
  __shared__ __align__(16) __bf16 Ps[4][16][72];
  __shared__ float maskb[64];

  const size_t headBase = ((size_t)(bz * NH_ + nh)) * S_ * DH_;
  const char* gK = (const char*)(Kh + headBase);
  const char* gV = (const char*)(Vh + headBase);
  const unsigned ksOff = (unsigned)(size_t)&Ks[0][0];
  const unsigned vsOff = (unsigned)(size_t)&Vs[0][0];

  { // stage Q tile once: 64x64 bf16 = 512 x 16B, async to LDS
    const char* gq = (const char*)(Qh + headBase + (size_t)q0 * DH_);
#pragma unroll
    for (int i = t; i < 512; i += 128) {
      int row = i >> 3, seg = i & 7;
      async_ld16(&Qs[row][seg * 8], gq + i * 16);
    }
  }
  wait_async0();
  __syncthreads();

  const v16bf a0 = load_A_frag(&Qs[w * 16][0],  72, lane);
  const v16bf a1 = load_A_frag(&Qs[w * 16][32], 72, lane);

  const v8f vzero = {0.f,0.f,0.f,0.f,0.f,0.f,0.f,0.f};
  v8f o[4];
  float row_max[8], row_sum[8];
#pragma unroll
  for (int i = 0; i < 4; ++i) o[i] = vzero;
#pragma unroll
  for (int r = 0; r < 8; ++r) { row_max[r] = -1e30f; row_sum[r] = 0.f; }

  for (int kt = 0; kt < S_ / 64; ++kt) {
    const int k0 = kt * 64;
    __syncthreads();  // all waves done reading previous Ks/Vs
    if (w == 0) {     // wave 0 drives the TDM for both tiles
      tdm_load_64x64_bf16(ksOff, gK + (size_t)k0 * DH_ * 2);
      tdm_load_64x64_bf16(vsOff, gV + (size_t)k0 * DH_ * 2);
      __builtin_amdgcn_s_wait_tensorcnt(0);
    }
    if (t < 64) maskb[t] = (1.0f - amask[bz * S_ + k0 + t]) * -10000.0f;
    __syncthreads(); // TDM data + mask visible to all waves

    // prefetch next tile into L2 (speculative; 128B line per thread)
    if (kt + 1 < S_ / 64) {
      const char* nk = gK + (size_t)(k0 + 64) * DH_ * 2;
      const char* nv = gV + (size_t)(k0 + 64) * DH_ * 2;
      if (t < 64) __builtin_prefetch(nk + t * 128, 0, 1);
      else        __builtin_prefetch(nv + (t - 64) * 128, 0, 1);
    }

    // scores: S = Q(16x64) * K^T(64x64); 4 n-frags x 2 k-steps
    v8f sc[4];
#pragma unroll
    for (int i = 0; i < 4; ++i) sc[i] = vzero;
#pragma unroll
    for (int wn = 0; wn < 4; ++wn) {
      v16bf bk0 = load_Bt_frag(&Ks[wn * 16][0],  72, lane);
      v16bf bk1 = load_Bt_frag(&Ks[wn * 16][32], 72, lane);
      sc[wn] = wmma_bf16(a0, bk0, sc[wn]);
      sc[wn] = wmma_bf16(a1, bk1, sc[wn]);
    }

    // online softmax across this key tile (rows live in 16-lane halves)
#pragma unroll
    for (int r = 0; r < 8; ++r) {
      float pv[4];
      float mloc = -1e30f;
#pragma unroll
      for (int wn = 0; wn < 4; ++wn) {
        float v = sc[wn][r] * 0.125f + maskb[wn * 16 + nlo];
        pv[wn] = v;
        mloc = fmaxf(mloc, v);
      }
#pragma unroll
      for (int off = 1; off < 16; off <<= 1)
        mloc = fmaxf(mloc, __shfl_xor(mloc, off, 32));
      float newm  = fmaxf(row_max[r], mloc);
      float alpha = __expf(row_max[r] - newm);
      row_max[r] = newm;
      row_sum[r] *= alpha;
#pragma unroll
      for (int wn = 0; wn < 4; ++wn) o[wn][r] *= alpha;
      float lsum = 0.f;
#pragma unroll
      for (int wn = 0; wn < 4; ++wn) {
        float p = __expf(pv[wn] - newm);
        lsum += p;
        Ps[w][r + hf * 8][wn * 16 + nlo] = (__bf16)p;   // C-layout -> LDS
      }
#pragma unroll
      for (int off = 1; off < 16; off <<= 1)
        lsum += __shfl_xor(lsum, off, 32);
      row_sum[r] += lsum;
    }
    // Ps is per-wave; wave LDS ops are program-ordered (DScnt) -> no barrier

    // ctx += P(16x64) * V(64x64)
    v16bf p0 = load_A_frag(&Ps[w][0][0],  72, lane);
    v16bf p1 = load_A_frag(&Ps[w][0][32], 72, lane);
#pragma unroll
    for (int wn = 0; wn < 4; ++wn) {
      v16bf bv0 = load_B_frag(&Vs[0][wn * 16],  72, lane);
      v16bf bv1 = load_B_frag(&Vs[32][wn * 16], 72, lane);
      o[wn] = wmma_bf16(p0, bv0, o[wn]);
      o[wn] = wmma_bf16(p1, bv1, o[wn]);
    }
  }

  // normalize + store fp32 in [B, S, H] layout
#pragma unroll
  for (int r = 0; r < 8; ++r) {
    int q = q0 + w * 16 + r + hf * 8;
    float inv = 1.0f / row_sum[r];
#pragma unroll
    for (int wn = 0; wn < 4; ++wn) {
      out[((size_t)(bz * S_ + q)) * H_ + nh * DH_ + wn * 16 + nlo] = o[wn][r] * inv;
    }
  }
}

// ---------------------------------------------------------------------------
extern "C" void kernel_launch(void* const* d_in, const int* in_sizes, int n_in,
                              void* d_out, int out_size, void* d_ws, size_t ws_size,
                              hipStream_t stream) {
  const float* X  = (const float*)d_in[0];
  const float* am = (const float*)d_in[1];
  const float* Wq = (const float*)d_in[2];
  const float* bq = (const float*)d_in[3];
  const float* Wk = (const float*)d_in[4];
  const float* bk = (const float*)d_in[5];
  const float* Wv = (const float*)d_in[6];
  const float* bv = (const float*)d_in[7];
  float* out = (float*)d_out;

  const size_t headElems = (size_t)B_ * NH_ * S_ * DH_;  // 4M elems -> 8MB bf16
  __bf16* Qh = (__bf16*)d_ws;
  __bf16* Kh = Qh + headElems;
  __bf16* Vh = Kh + headElems;

  dim3 g1(H_ / 64, (B_ * S_) / 128, 3);
  qkv_kernel<<<g1, dim3(256), 0, stream>>>(X, Wq, bq, Wk, bk, Wv, bv, Qh, Kh, Vh);

  dim3 g2(S_ / 64, NH_, B_);
  attn_kernel<<<g2, dim3(128), 0, stream>>>(Qh, Kh, Vh, am, out);
}
